// Model_9405978378818
// MI455X (gfx1250) — matrix operand order repaired and checked by reading, not compiled, not run on hardware
//
#include <hip/hip_runtime.h>

// ---------------- WMMA types & helpers (CDNA5 gfx1250, wave32) ----------------
typedef __attribute__((ext_vector_type(16))) __bf16 v16bf;
typedef __attribute__((ext_vector_type(8)))  float  v8f;

union BF16x16 { unsigned short u[16]; v16bf v; };

__device__ __forceinline__ v8f wmma_bf16(const v16bf& a, const v16bf& b, const v8f& c) {
    return __builtin_amdgcn_wmma_f32_16x16x32_bf16(false, a, false, b, (short)0, c, false, false);
}

__device__ __forceinline__ unsigned short f2bf(float f) {
    unsigned int u = __float_as_uint(f);
    unsigned int r = u + 0x7FFFu + ((u >> 16) & 1u);   // round-to-nearest-even
    return (unsigned short)(r >> 16);
}

// A-matrix (16x32 bf16) lane layout: lane<16: elems0..15 -> K0..15
//                                    lane>=16: elems0..15 -> K16..31
__device__ __forceinline__ int a_k(int lane, int i) {
    int base = ((lane >> 4) & 1) * 16;
    return base + i;
}
// B-matrix (32x16 bf16): lanes0-15 hold K0..15, lanes16-31 hold K16..31 (col = lane&15)
__device__ __forceinline__ int b_k0(int lane) { return ((lane >> 4) & 1) * 16; }
// C/D (16x16 f32): elem r -> row = r + 8*(lane>=16), col = lane&15
__device__ __forceinline__ int c_row(int lane, int r) { return r + ((lane >> 4) & 1) * 8; }

__device__ __forceinline__ float siluf(float a) { return a / (1.f + __expf(-a)); }

// ---------------- prep: fp32 -> bf16 weight conversion ----------------
__global__ void k_f2bf(const float* __restrict__ src, unsigned short* __restrict__ dst, int n) {
    int i = blockIdx.x * 256 + threadIdx.x;
    if (i < n) dst[i] = f2bf(src[i]);
}

// conv_w (512,32,3) -> wcb[d][k], k = kk*32 + c  (K-contiguous per output channel)
__global__ void k_convw(const float* __restrict__ cw, unsigned short* __restrict__ dst) {
    int i = blockIdx.x * 256 + threadIdx.x;
    if (i < 512 * 96) {
        int d = i / 96, k = i % 96, kk = k / 32, c = k % 32;
        dst[i] = f2bf(cw[(d * 32 + c) * 3 + kk]);
    }
}

// ---------------- K1: per-(b,c) mean/std over L, write normalized x ----------------
__global__ __launch_bounds__(256) void k_norm(const float* __restrict__ x,
                                              float* __restrict__ meanb, float* __restrict__ stdb,
                                              float* __restrict__ xn) {
    int bc = blockIdx.x;             // B*C = 1024
    int b = bc >> 5, c = bc & 31;
    int tid = threadIdx.x;
    __shared__ float s1[256], s2[256];
    float sum = 0.f, sq = 0.f;
    for (int l = tid; l < 1024; l += 256) {
        float v = x[((size_t)b * 1024 + l) * 32 + c];
        sum += v; sq += v * v;
    }
    s1[tid] = sum; s2[tid] = sq; __syncthreads();
    for (int off = 128; off > 0; off >>= 1) {
        if (tid < off) { s1[tid] += s1[tid + off]; s2[tid] += s2[tid + off]; }
        __syncthreads();
    }
    __shared__ float mu, istd;
    if (tid == 0) {
        float m = s1[0] * (1.f / 1024.f);
        float var = s2[0] * (1.f / 1024.f) - m * m;
        float st = sqrtf(var + 1e-5f);
        meanb[bc] = m; stdb[bc] = st;
        mu = m; istd = 1.f / st;
    }
    __syncthreads();
    for (int l = tid; l < 1024; l += 256)
        xn[((size_t)b * 1024 + l) * 32 + c] = (x[((size_t)b * 1024 + l) * 32 + c] - mu) * istd;
}

// compact row set: m-tiles 0..6 -> l0 = 912+16*mt ; m-tile 7 -> l0 = 0
__device__ __forceinline__ int tile_l0(int mt) { return (mt < 7) ? (912 + mt * 16) : 0; }
__device__ __forceinline__ size_t xsc_off(int b, int lp) {   // row base in compact xs buffer
    int mt, m;
    if (lp >= 912) { mt = (lp - 912) >> 4; m = (lp - 912) & 15; }
    else           { mt = 7;               m = lp; }
    return (((size_t)b * 8 + mt) * 16 + m) * 32;
}

// ---------------- K2: e1 = circConv3(xn) + mark@temp_w^T + PE  (only live tiles) ----------------
__global__ __launch_bounds__(256) void k_e1(const float* __restrict__ xn,
                                            const unsigned short* __restrict__ wcb,
                                            const float* __restrict__ xmark,
                                            const float* __restrict__ temp_w,
                                            float* __restrict__ e1c) {
    int mt = blockIdx.x, b = blockIdx.y;
    int tid = threadIdx.x, wave = tid >> 5, lane = tid & 31;
    int l0 = tile_l0(mt);
    int l = l0 + (lane & 15);
    BF16x16 afr[3];
    for (int kt = 0; kt < 3; ++kt) {
        int lsrc = (l + kt - 1 + 1024) & 1023;          // circular pad
        const float* src = xn + ((size_t)b * 1024 + lsrc) * 32;
        for (int i = 0; i < 16; ++i) afr[kt].u[i] = f2bf(src[a_k(lane, i)]);
    }
    // hoist the 8 mark float4 loads (same rows for every n-tile)
    float4 mk[8];
    for (int r = 0; r < 8; ++r) {
        int lm = l0 + c_row(lane, r);
        mk[r] = *(const float4*)(xmark + ((size_t)b * 1024 + lm) * 4);
    }
    for (int j = 0; j < 4; ++j) {
        int d0 = (wave * 4 + j) * 16;
        int d = d0 + (lane & 15);
        v8f acc = {};
        for (int kt = 0; kt < 3; ++kt) {
            BF16x16 bf;
            const unsigned short* s = wcb + (size_t)d * 96 + kt * 32 + b_k0(lane);
            for (int i = 0; i < 16; ++i) bf.u[i] = s[i];
            acc = wmma_bf16(afr[kt].v, bf.v, acc);
        }
        float tw0 = temp_w[d * 4 + 0], tw1 = temp_w[d * 4 + 1];
        float tw2 = temp_w[d * 4 + 2], tw3 = temp_w[d * 4 + 3];
        float div = __expf((float)(2 * (d >> 1)) * (-9.210340371976184f / 512.f));
        float ph  = (d & 1) ? 1.5707963267948966f : 0.f;   // cos(x)=sin(x+pi/2): no divergence
        for (int r = 0; r < 8; ++r) {
            int m = c_row(lane, r);
            int lm = l0 + m;
            float extra = mk[r].x * tw0 + mk[r].y * tw1 + mk[r].z * tw2 + mk[r].w * tw3;
            extra += __sinf((float)lm * div + ph);
            e1c[(((size_t)b * 8 + mt) * 16 + m) * 512 + d] = acc[r] + extra;
        }
    }
}

// ---------------- K3: scale = exp(e1 @ hetero_w^T + b); xs = xn*scale ----------------
__global__ __launch_bounds__(64) void k_scale(const float* __restrict__ e1c,
                                              const unsigned short* __restrict__ hwb,
                                              const float* __restrict__ hb,
                                              const float* __restrict__ xn,
                                              float* __restrict__ xsc) {
    int mt = blockIdx.x, b = blockIdx.y;
    int tid = threadIdx.x, wave = tid >> 5, lane = tid & 31;
    int l0 = tile_l0(mt);
    const float* arow = e1c + (((size_t)b * 8 + mt) * 16 + (lane & 15)) * 512;
    int c = wave * 16 + (lane & 15);
    v8f acc = {};
    for (int kt = 0; kt < 16; ++kt) {
        BF16x16 a, bb;
        for (int i = 0; i < 16; ++i) a.u[i] = f2bf(arow[kt * 32 + a_k(lane, i)]);
        const unsigned short* s = hwb + (size_t)c * 512 + kt * 32 + b_k0(lane);
        for (int i = 0; i < 16; ++i) bb.u[i] = s[i];
        acc = wmma_bf16(a.v, bb.v, acc);
    }
    float bias = hb[c];
    for (int r = 0; r < 8; ++r) {
        int m = c_row(lane, r);
        int lm = l0 + m;
        float sc = __expf(acc[r] + bias);
        float xv = xn[((size_t)b * 1024 + lm) * 32 + c];
        xsc[(((size_t)b * 8 + mt) * 16 + m) * 32 + c] = xv * sc;
    }
}

// ---------------- K4: e2 from xs; xc = e1+e2 -> bf16 in (g, b*6+p, 16, 128) layout ----------------
__global__ __launch_bounds__(256) void k_e2xc(const float* __restrict__ xscb,
                                              const unsigned short* __restrict__ wcb,
                                              const float* __restrict__ xmark,
                                              const float* __restrict__ temp_w,
                                              const float* __restrict__ e1c,
                                              unsigned short* __restrict__ xcbf) {
    int pt = blockIdx.x, b = blockIdx.y;        // pt in 0..5 -> patches 58..63
    int tid = threadIdx.x, wave = tid >> 5, lane = tid & 31;
    int l0 = 928 + pt * 16;
    int l = l0 + (lane & 15);
    BF16x16 afr[3];
    for (int kt = 0; kt < 3; ++kt) {
        int lsrc = (l + kt - 1 + 1024) & 1023;
        const float* src = xscb + xsc_off(b, lsrc);
        for (int i = 0; i < 16; ++i) afr[kt].u[i] = f2bf(src[a_k(lane, i)]);
    }
    float4 mk[8];
    for (int r = 0; r < 8; ++r) {
        int lm = l0 + c_row(lane, r);
        mk[r] = *(const float4*)(xmark + ((size_t)b * 1024 + lm) * 4);
    }
    int e1mt = 1 + pt;
    for (int j = 0; j < 4; ++j) {
        int d0 = (wave * 4 + j) * 16;
        int d = d0 + (lane & 15);
        v8f acc = {};
        for (int kt = 0; kt < 3; ++kt) {
            BF16x16 bf;
            const unsigned short* s = wcb + (size_t)d * 96 + kt * 32 + b_k0(lane);
            for (int i = 0; i < 16; ++i) bf.u[i] = s[i];
            acc = wmma_bf16(afr[kt].v, bf.v, acc);
        }
        float tw0 = temp_w[d * 4 + 0], tw1 = temp_w[d * 4 + 1];
        float tw2 = temp_w[d * 4 + 2], tw3 = temp_w[d * 4 + 3];
        float div = __expf((float)(2 * (d >> 1)) * (-9.210340371976184f / 512.f));
        float ph  = (d & 1) ? 1.5707963267948966f : 0.f;
        int g = d >> 7, dg = d & 127;
        for (int r = 0; r < 8; ++r) {
            int m = c_row(lane, r);
            int lm = l0 + m;
            float extra = mk[r].x * tw0 + mk[r].y * tw1 + mk[r].z * tw2 + mk[r].w * tw3;
            extra += __sinf((float)lm * div + ph);
            float v = e1c[(((size_t)b * 8 + e1mt) * 16 + m) * 512 + d] + acc[r] + extra;
            xcbf[(((size_t)g * 192 + (b * 6 + pt)) * 16 + m) * 128 + dg] = f2bf(v);
        }
    }
}

// ---------------- K5: fused Mamba block per (group, patch) ----------------
__global__ __launch_bounds__(256) void k_mamba(const unsigned short* __restrict__ xcbf,
                                               const unsigned short* __restrict__ inwb,
                                               const float* __restrict__ conv1_w,
                                               const float* __restrict__ conv1_b,
                                               const float* __restrict__ xproj_w,
                                               const float* __restrict__ dt_w,
                                               const float* __restrict__ dt_b,
                                               const float* __restrict__ A_log,
                                               const float* __restrict__ Dp,
                                               const unsigned short* __restrict__ outwb,
                                               float* __restrict__ xmc) {
    __shared__ float xz[16][512];    // in_proj out: [:, :256]=conv in / later y, [:,256:]=z
    __shared__ float xcv[16][256];   // conv+silu result
    __shared__ float dbl[16][48];    // x_proj out (dt_raw | B | C)
    int bp = blockIdx.x;             // 192 live patches
    int g  = blockIdx.y;             // 4 groups
    int tid = threadIdx.x, wave = tid >> 5, lane = tid & 31;

    // pull the weight panels toward this WGP while fragments are assembled
    __builtin_prefetch(inwb  + ((size_t)g * 512 + wave * 64) * 128, 0, 1);
    __builtin_prefetch(outwb + ((size_t)g * 128 + wave * 16) * 256, 0, 1);

    // --- in_proj: (16x128) x (128x512) via WMMA bf16 ---
    BF16x16 afr[4];
    {
        const unsigned short* arow = xcbf + (((size_t)g * 192 + bp) * 16 + (lane & 15)) * 128;
        for (int kt = 0; kt < 4; ++kt)
            for (int i = 0; i < 16; ++i) afr[kt].u[i] = arow[kt * 32 + a_k(lane, i)];
    }
    for (int j = 0; j < 4; ++j) {
        int e = (wave * 4 + j) * 16 + (lane & 15);
        v8f acc = {};
        const unsigned short* brow = inwb + ((size_t)g * 512 + e) * 128;
        for (int kt = 0; kt < 4; ++kt) {
            BF16x16 bf;
            const unsigned short* s = brow + kt * 32 + b_k0(lane);
            for (int i = 0; i < 16; ++i) bf.u[i] = s[i];
            acc = wmma_bf16(afr[kt].v, bf.v, acc);
        }
        for (int r = 0; r < 8; ++r) xz[c_row(lane, r)][e] = acc[r];
    }
    __syncthreads();

    // --- depthwise conv (k=4, causal) + silu : thread = channel e ---
    {
        int e = tid;
        float w0 = conv1_w[((size_t)g * 256 + e) * 4 + 0];
        float w1 = conv1_w[((size_t)g * 256 + e) * 4 + 1];
        float w2 = conv1_w[((size_t)g * 256 + e) * 4 + 2];
        float w3 = conv1_w[((size_t)g * 256 + e) * 4 + 3];
        float cb = conv1_b[g * 256 + e];
        for (int t = 0; t < 16; ++t) {
            float a = cb + xz[t][e] * w3;
            if (t >= 1) a += xz[t - 1][e] * w2;
            if (t >= 2) a += xz[t - 2][e] * w1;
            if (t >= 3) a += xz[t - 3][e] * w0;
            xcv[t][e] = siluf(a);
        }
    }
    __syncthreads();

    // --- x_proj: dbl[t][j] = sum_e xcv[t][e] * xp_w[j][e]   (16x40, K=256) ---
    for (int o = tid; o < 640; o += 256) {
        int t = o / 40, jj = o % 40;
        const float* w = xproj_w + ((size_t)g * 40 + jj) * 256;
        float a = 0.f;
        for (int e = 0; e < 256; ++e) a += xcv[t][e] * w[e];
        dbl[t][jj] = a;
    }
    __syncthreads();

    // --- selective scan: thread = channel e, state h[16] in regs ---
    {
        int e = tid;
        float Av[16];
        for (int n = 0; n < 16; ++n) Av[n] = -__expf(A_log[((size_t)g * 256 + e) * 16 + n]);
        float wr[8];
        for (int r = 0; r < 8; ++r) wr[r] = dt_w[((size_t)g * 256 + e) * 8 + r];
        float dbias = dt_b[g * 256 + e];
        float Dv = Dp[g * 256 + e];
        float h[16];
        for (int n = 0; n < 16; ++n) h[n] = 0.f;
        for (int t = 0; t < 16; ++t) {
            float raw = dbias;
            for (int r = 0; r < 8; ++r) raw += dbl[t][r] * wr[r];
            float dtv = (raw > 20.f) ? raw : log1pf(__expf(raw));   // softplus
            float xv = xcv[t][e];
            float acc = 0.f;
            for (int n = 0; n < 16; ++n) {
                h[n] = __expf(dtv * Av[n]) * h[n] + dtv * dbl[t][8 + n] * xv;
                acc += h[n] * dbl[t][24 + n];
            }
            float y = acc + Dv * xv;
            y *= siluf(xz[t][256 + e]);     // gate
            xz[t][e] = y;                    // overwrite (first half dead now)
        }
    }
    __syncthreads();

    // --- out_proj: (16x256) x (256x128) WMMA; wave w -> dg tile w ---
    {
        BF16x16 a2[8];
        int row = lane & 15;
        for (int kt = 0; kt < 8; ++kt)
            for (int i = 0; i < 16; ++i) a2[kt].u[i] = f2bf(xz[row][kt * 32 + a_k(lane, i)]);
        int dg = wave * 16 + (lane & 15);
        const unsigned short* brow = outwb + ((size_t)g * 128 + dg) * 256;
        v8f acc = {};
        for (int kt = 0; kt < 8; ++kt) {
            BF16x16 bf;
            const unsigned short* s = brow + kt * 32 + b_k0(lane);
            for (int i = 0; i < 16; ++i) bf.u[i] = s[i];
            acc = wmma_bf16(a2[kt].v, bf.v, acc);
        }
        for (int r = 0; r < 8; ++r) {
            int m = c_row(lane, r);
            xmc[((size_t)bp * 16 + m) * 512 + g * 128 + dg] = acc[r];
        }
    }
}

// ---------------- K6: head GEMM (512->32) + de-normalize ----------------
__global__ __launch_bounds__(64) void k_head(const float* __restrict__ xmc,
                                             const unsigned short* __restrict__ owwb,
                                             const float* __restrict__ meanb,
                                             const float* __restrict__ stdb,
                                             float* __restrict__ out) {
    int mt = blockIdx.x;                    // 192 row tiles (3072 rows = B*96)
    int tid = threadIdx.x, wave = tid >> 5, lane = tid & 31;
    const float* arow = xmc + ((size_t)mt * 16 + (lane & 15)) * 512;
    int c = wave * 16 + (lane & 15);
    v8f acc = {};
    for (int kt = 0; kt < 16; ++kt) {
        BF16x16 a, bb;
        for (int i = 0; i < 16; ++i) a.u[i] = f2bf(arow[kt * 32 + a_k(lane, i)]);
        const unsigned short* s = owwb + (size_t)c * 512 + kt * 32 + b_k0(lane);
        for (int i = 0; i < 16; ++i) bb.u[i] = s[i];
        acc = wmma_bf16(a.v, bb.v, acc);
    }
    for (int r = 0; r < 8; ++r) {
        int grow = mt * 16 + c_row(lane, r);
        int b = grow / 96, lr = grow % 96;
        out[((size_t)b * 96 + lr) * 32 + c] = acc[r] * stdb[b * 32 + c] + meanb[b * 32 + c];
    }
}

// ---------------- launch ----------------
extern "C" void kernel_launch(void* const* d_in, const int* in_sizes, int n_in,
                              void* d_out, int out_size, void* d_ws, size_t ws_size,
                              hipStream_t stream) {
    const float* x_enc    = (const float*)d_in[0];
    const float* x_mark   = (const float*)d_in[1];
    const float* conv_w   = (const float*)d_in[4];
    const float* temp_w   = (const float*)d_in[5];
    const float* hetero_w = (const float*)d_in[6];
    const float* hetero_b = (const float*)d_in[7];
    const float* in_proj  = (const float*)d_in[8];
    const float* conv1_w  = (const float*)d_in[9];
    const float* conv1_b  = (const float*)d_in[10];
    const float* x_proj   = (const float*)d_in[11];
    const float* dt_w     = (const float*)d_in[12];
    const float* dt_b     = (const float*)d_in[13];
    const float* A_log    = (const float*)d_in[14];
    const float* D_param  = (const float*)d_in[15];
    const float* out_proj = (const float*)d_in[16];
    const float* out_w    = (const float*)d_in[17];
    float* out = (float*)d_out;

    char* p = (char*)d_ws;
    auto alloc_f = [&](size_t n) { float* r = (float*)p; p += ((n * 4 + 255) / 256) * 256; return r; };
    auto alloc_u = [&](size_t n) { unsigned short* r = (unsigned short*)p; p += ((n * 2 + 255) / 256) * 256; return r; };

    float* meanb = alloc_f(1024);
    float* stdb  = alloc_f(1024);
    float* xn    = alloc_f((size_t)32 * 1024 * 32);
    float* e1c   = alloc_f((size_t)32 * 8 * 16 * 512);
    float* xscb  = alloc_f((size_t)32 * 8 * 16 * 32);
    float* xmc   = alloc_f((size_t)192 * 16 * 512);
    unsigned short* xcbf  = alloc_u((size_t)4 * 192 * 16 * 128);
    unsigned short* wcb   = alloc_u(512 * 96);
    unsigned short* hwb   = alloc_u(32 * 512);
    unsigned short* inwb  = alloc_u((size_t)4 * 512 * 128);
    unsigned short* outwb = alloc_u((size_t)4 * 128 * 256);
    unsigned short* owwb  = alloc_u(32 * 512);
    (void)ws_size; (void)in_sizes; (void)n_in; (void)out_size;

    k_f2bf<<<(262144 + 255) / 256, 256, 0, stream>>>(in_proj, inwb, 262144);
    k_f2bf<<<(131072 + 255) / 256, 256, 0, stream>>>(out_proj, outwb, 131072);
    k_f2bf<<<64, 256, 0, stream>>>(hetero_w, hwb, 16384);
    k_f2bf<<<64, 256, 0, stream>>>(out_w, owwb, 16384);
    k_convw<<<192, 256, 0, stream>>>(conv_w, wcb);

    k_norm<<<1024, 256, 0, stream>>>(x_enc, meanb, stdb, xn);
    k_e1<<<dim3(8, 32), 256, 0, stream>>>(xn, wcb, x_mark, temp_w, e1c);
    k_scale<<<dim3(8, 32), 64, 0, stream>>>(e1c, hwb, hetero_b, xn, xscb);
    k_e2xc<<<dim3(6, 32), 256, 0, stream>>>(xscb, wcb, x_mark, temp_w, e1c, xcbf);
    k_mamba<<<dim3(192, 4), 256, 0, stream>>>(xcbf, inwb, conv1_w, conv1_b, x_proj,
                                              dt_w, dt_b, A_log, D_param, outwb, xmc);
    k_head<<<192, 64, 0, stream>>>(xmc, owwb, meanb, stdb, out);
}
